// FusedLlamaLowBitMultiDecoderlayer_58884001628619
// MI455X (gfx1250) — matrix operand order repaired and checked by reading, not compiled, not run on hardware
//
#include <hip/hip_runtime.h>
#include <hip/hip_bf16.h>

#define S_   2048
#define H_   2048
#define NH_  16
#define D_   128
#define I_   5632
#define L_   2

typedef __attribute__((ext_vector_type(16))) _Float16 v16h;
typedef __attribute__((ext_vector_type(8)))  _Float16 v8h;
typedef __attribute__((ext_vector_type(8)))  float    v8f;
typedef int gv4i __attribute__((vector_size(16)));   // matches builtin's int4 pointee

#define WMMA_F16(a,b,c) __builtin_amdgcn_wmma_f32_16x16x32_f16(false,(a),false,(b),(short)0,(c),false,false)

#if __has_builtin(__builtin_amdgcn_global_load_async_to_lds_b128)
#define USE_ASYNC 1
#else
#define USE_ASYNC 0
#endif

#if USE_ASYNC
#if __has_builtin(__builtin_amdgcn_s_wait_asynccnt)
#define WAIT_ASYNC(n) __builtin_amdgcn_s_wait_asynccnt(n)
#else
#define WAIT_ASYNC(n) asm volatile("s_wait_asynccnt %0" :: "i"(n) : "memory")
#endif
__device__ __forceinline__ void async_cp16(const _Float16* g, _Float16* l) {
  __builtin_amdgcn_global_load_async_to_lds_b128(
      (__attribute__((address_space(1))) gv4i*)g,
      (__attribute__((address_space(3))) gv4i*)l, 0, 0);
}
#else
#define WAIT_ASYNC(n) ((void)0)
#endif

// Build a v16h fragment from two 16-byte chunks (ISA 7.12.2 A/B layouts).
__device__ __forceinline__ v16h load_frag2(const _Float16* p0, const _Float16* p1) {
  v16h r;
  ((v8h*)&r)[0] = *(const v8h*)p0;
  ((v8h*)&r)[1] = *(const v8h*)p1;
  return r;
}

// ---------------------------------------------------------------- cast f32->f16
__global__ void __launch_bounds__(256) cast_f16_kernel(const float* __restrict__ in,
                                                       _Float16* __restrict__ out, size_t n) {
  size_t i = (size_t)blockIdx.x * blockDim.x + threadIdx.x;
  size_t stride = (size_t)gridDim.x * blockDim.x;
  for (; i < n; i += stride) out[i] = (_Float16)in[i];
}

// ---------------------------------------------------------------- RMSNorm -> f16
__global__ void __launch_bounds__(256) rmsnorm_kernel(const float* __restrict__ x,
                                                      const float* __restrict__ w,
                                                      _Float16* __restrict__ out) {
  __shared__ float red[256];
  int row = blockIdx.x;
  int tid = threadIdx.x;
  const float* xr = x + (size_t)row * H_;
  float s = 0.f;
  for (int j = tid; j < H_; j += 256) { float v = xr[j]; s += v * v; }
  red[tid] = s;
  __syncthreads();
  for (int k = 128; k > 0; k >>= 1) {
    if (tid < k) red[tid] += red[tid + k];
    __syncthreads();
  }
  float rstd = rsqrtf(red[0] * (1.0f / H_) + 1e-6f);
  for (int j = tid; j < H_; j += 256)
    out[(size_t)row * H_ + j] = (_Float16)(xr[j] * rstd * w[j]);
}

// ---------------------------------------------------------------- RoPE on q and k (in place, pairwise safe)
__global__ void __launch_bounds__(256) rope_kernel(_Float16* __restrict__ q,
                                                   _Float16* __restrict__ k,
                                                   const float* __restrict__ cosT,
                                                   const float* __restrict__ sinT) {
  int t = blockIdx.x * blockDim.x + threadIdx.x;   // S*NH*(D/2) threads
  int d  = t & 63;
  int hh = (t >> 6) & (NH_ - 1);
  int s  = t >> 10;
  float c  = cosT[s * D_ + d];     // tables repeat over both halves
  float sn = sinT[s * D_ + d];
  size_t base = (size_t)s * H_ + hh * D_ + d;
  float q1 = (float)q[base], q2 = (float)q[base + 64];
  q[base]      = (_Float16)(q1 * c - q2 * sn);
  q[base + 64] = (_Float16)(q2 * c + q1 * sn);
  float k1 = (float)k[base], k2 = (float)k[base + 64];
  k[base]      = (_Float16)(k1 * c - k2 * sn);
  k[base + 64] = (_Float16)(k2 * c + k1 * sn);
}

// ---------------------------------------------------------------- GEMM: C[M,N] = A[M,K] * W[N,K]^T
// modes: 0=f16 out, 1=f16 silu, 2=f16 out*aux16, 3=f32 out=acc+res32, 4=f16 transposed store (out[n*M+m])
#define BM 128
#define BN 128
#define BK 64
#define PADH 72   // halves per LDS row (144B = 36 dwords -> conflict-free b128 frag reads)

// Stage one thread's 32-half slice of a 128x64 tile into LDS.
__device__ __forceinline__ void stage_tile(const _Float16* __restrict__ g, int ld, int rowbase,
                                           int lr, int lc, int k0, _Float16* lrow) {
  const _Float16* gp = g + (size_t)(rowbase + lr) * ld + k0 + lc;
#if USE_ASYNC
  for (int j = 0; j < 4; ++j) async_cp16(gp + j * 8, lrow + j * 8);
#else
  const uint4* gv = (const uint4*)gp;
  uint4* lv = (uint4*)lrow;
  for (int j = 0; j < 4; ++j) lv[j] = gv[j];
#endif
}

__global__ void __launch_bounds__(256) gemm_f16_kernel(
    const _Float16* __restrict__ A, const _Float16* __restrict__ Bw, int K,
    _Float16* __restrict__ out16, float* __restrict__ out32,
    const _Float16* __restrict__ aux16, const float* __restrict__ res32, int mode) {
  __shared__ alignas(16) _Float16 sA[2][BM * PADH];
  __shared__ alignas(16) _Float16 sB[2][BN * PADH];

  int tid = threadIdx.x;
  int wv = tid >> 5, lane = tid & 31;
  int hi = lane >> 4, lo = lane & 15;
  int wm = wv >> 2, wn = wv & 3;          // 2x4 wave grid
  int m0 = blockIdx.y * BM, n0 = blockIdx.x * BN;
  int N = gridDim.x * BN, M = gridDim.y * BM;

  v8f acc[4][2] = {};

  int lr = tid >> 1;                      // 0..127 tile row
  int lc = (tid & 1) * 32;                // half-offset 0 or 32

  int nk = K / BK;
  // prologue: stage tile 0 into buffer 0
  stage_tile(A,  K, m0, lr, lc, 0, &sA[0][lr * PADH + lc]);
  stage_tile(Bw, K, n0, lr, lc, 0, &sB[0][lr * PADH + lc]);

  for (int kt = 0; kt < nk; ++kt) {
    int cur = kt & 1;
    if (kt + 1 < nk) {
      stage_tile(A,  K, m0, lr, lc, (kt + 1) * BK, &sA[cur ^ 1][lr * PADH + lc]);
      stage_tile(Bw, K, n0, lr, lc, (kt + 1) * BK, &sB[cur ^ 1][lr * PADH + lc]);
      WAIT_ASYNC(8);     // async loads complete in order: tile kt's 8 ops are done
    } else {
      WAIT_ASYNC(0);
    }
    __syncthreads();

    for (int kc = 0; kc < 2; ++kc) {
      v16h af[4], bf[2];
      for (int mi = 0; mi < 4; ++mi) {
        const _Float16* p = &sA[cur][(wm * 64 + mi * 16 + lo) * PADH + kc * 32];
        af[mi] = load_frag2(p + hi * 8, p + 16 + hi * 8);
      }
      for (int ni = 0; ni < 2; ++ni) {
        const _Float16* p = &sB[cur][(wn * 32 + ni * 16 + lo) * PADH + kc * 32];
        bf[ni] = load_frag2(p + hi * 8, p + 16 + hi * 8);
      }
      for (int mi = 0; mi < 4; ++mi)
        for (int ni = 0; ni < 2; ++ni)
          acc[mi][ni] = WMMA_F16(af[mi], bf[ni], acc[mi][ni]);
    }
    __syncthreads();   // all waves done reading buffer `cur` before it is re-staged
  }

  for (int mi = 0; mi < 4; ++mi)
    for (int ni = 0; ni < 2; ++ni)
      for (int r = 0; r < 8; ++r) {
        int m = m0 + wm * 64 + mi * 16 + hi * 8 + r;
        int n = n0 + wn * 32 + ni * 16 + lo;
        float v = acc[mi][ni][r];
        size_t idx = (size_t)m * N + n;
        switch (mode) {
          case 0: out16[idx] = (_Float16)v; break;
          case 1: out16[idx] = (_Float16)(v / (1.f + __expf(-v))); break;               // SiLU
          case 2: out16[idx] = (_Float16)(v * (float)aux16[idx]); break;                // * gate
          case 3: out32[idx] = v + res32[idx]; break;                                   // + residual
          case 4: out16[(size_t)n * M + m] = (_Float16)v; break;                        // V^T store
        }
      }
}

// ---------------------------------------------------------------- flash attention
// One wave per (head, 16-query-row tile); streams 32 keys per iteration.
// q,k: [S,H] f16 (RoPE applied). vT: [H,S] f16. out: [S,H] f16.
__global__ void __launch_bounds__(128) flash_attn_kernel(
    const _Float16* __restrict__ q, const _Float16* __restrict__ k,
    const _Float16* __restrict__ vT, _Float16* __restrict__ o) {
  __shared__ alignas(16) _Float16 sP[4 * 16 * PADH];   // per-wave 16x32 P staging
  int wv = threadIdx.x >> 5, lane = threadIdx.x & 31;
  int hi = lane >> 4, lo = lane & 15;
  int wgid = blockIdx.x * 4 + wv;
  int h  = wgid >> 7;                 // / (S/16)
  int qt = wgid & 127;
  _Float16* P = sP + wv * 16 * PADH;

  // Q fragments: 16x128 split into 4 A-frags of 16x32
  v16h qf[4];
  const _Float16* qrow = q + (size_t)(qt * 16 + lo) * H_ + h * D_;
  for (int c = 0; c < 4; ++c)
    qf[c] = load_frag2(qrow + c * 32 + hi * 8, qrow + c * 32 + 16 + hi * 8);

  v8f acc[8] = {};
  float mrow[8], lrow[8];
  for (int r = 0; r < 8; ++r) { mrow[r] = -1e30f; lrow[r] = 0.f; }
  const float scale = 0.08838834764831845f;   // 1/sqrt(128)

  int kbEnd = (qt * 16 + 15) >> 5;
  for (int kb = 0; kb <= kbEnd; ++kb) {
    int key0 = kb * 32;
    v8f s0 = {}, s1 = {};
    const _Float16* k0p = k + (size_t)(key0 + lo) * H_ + h * D_;        // this lane's key column
    const _Float16* k1p = k0p + (size_t)16 * H_;
    for (int c = 0; c < 4; ++c) {
      v16h kf0 = load_frag2(k0p + c * 32 + hi * 8, k0p + c * 32 + 16 + hi * 8);
      v16h kf1 = load_frag2(k1p + c * 32 + hi * 8, k1p + c * 32 + 16 + hi * 8);
      s0 = WMMA_F16(qf[c], kf0, s0);
      s1 = WMMA_F16(qf[c], kf1, s1);
    }
    // online softmax over the 32 new columns
    for (int r = 0; r < 8; ++r) {
      int m  = qt * 16 + hi * 8 + r;
      int n0 = key0 + lo, n1 = n0 + 16;
      float a0 = s0[r] * scale + (n0 > m ? -1e9f : 0.f);
      float a1 = s1[r] * scale + (n1 > m ? -1e9f : 0.f);
      float t = fmaxf(a0, a1);
      t = fmaxf(t, __shfl_xor(t, 1));
      t = fmaxf(t, __shfl_xor(t, 2));
      t = fmaxf(t, __shfl_xor(t, 4));
      t = fmaxf(t, __shfl_xor(t, 8));
      float newm = fmaxf(mrow[r], t);
      float sc = __expf(mrow[r] - newm);
      float p0 = __expf(a0 - newm);
      float p1 = __expf(a1 - newm);
      float ps = p0 + p1;
      ps += __shfl_xor(ps, 1); ps += __shfl_xor(ps, 2);
      ps += __shfl_xor(ps, 4); ps += __shfl_xor(ps, 8);
      lrow[r] = lrow[r] * sc + ps;
      mrow[r] = newm;
      for (int dc = 0; dc < 8; ++dc) acc[dc][r] *= sc;   // rescale O rows
      int prow = hi * 8 + r;
      P[prow * PADH + lo]      = (_Float16)p0;
      P[prow * PADH + lo + 16] = (_Float16)p1;
    }
    // O += P(16x32) x V(32x128): reload P in A layout from LDS
    v16h pf = load_frag2(P + lo * PADH + hi * 8, P + lo * PADH + 16 + hi * 8);
    for (int dc = 0; dc < 8; ++dc) {
      const _Float16* vp = vT + (size_t)(h * D_ + dc * 16 + lo) * S_ + key0;
      v16h vf = load_frag2(vp + hi * 8, vp + 16 + hi * 8);
      acc[dc] = WMMA_F16(pf, vf, acc[dc]);
    }
  }

  for (int dc = 0; dc < 8; ++dc)
    for (int r = 0; r < 8; ++r) {
      int m = qt * 16 + hi * 8 + r;
      o[(size_t)m * H_ + h * D_ + dc * 16 + lo] = (_Float16)(acc[dc][r] / lrow[r]);
    }
}

// ---------------------------------------------------------------- host
extern "C" void kernel_launch(void* const* d_in, const int* in_sizes, int n_in,
                              void* d_out, int out_size, void* d_ws, size_t ws_size,
                              hipStream_t stream) {
  const float* hs   = (const float*)d_in[0];
  const float* cosT = (const float*)d_in[2];
  const float* sinT = (const float*)d_in[3];
  const float* wq   = (const float*)d_in[4];
  const float* wk   = (const float*)d_in[5];
  const float* wv   = (const float*)d_in[6];
  const float* wo   = (const float*)d_in[7];
  const float* wg   = (const float*)d_in[8];
  const float* wu   = (const float*)d_in[9];
  const float* wd   = (const float*)d_in[10];
  const float* ln1  = (const float*)d_in[11];
  const float* ln2  = (const float*)d_in[12];

  const size_t SH  = (size_t)S_ * H_;
  const size_t SI  = (size_t)S_ * I_;
  const size_t NQ  = (size_t)L_ * H_ * H_;
  const size_t NGU = (size_t)L_ * I_ * H_;

  char* ws = (char*)d_ws;
  size_t off = 0;
  auto alloc = [&](size_t bytes) -> char* {
    char* p = ws + off;
    off += (bytes + 255) & ~(size_t)255;
    return p;
  };
  float*     x    = (float*)    alloc(SH * 4);
  _Float16*  h16  = (_Float16*) alloc(SH * 2);
  _Float16*  q16  = (_Float16*) alloc(SH * 2);
  _Float16*  k16  = (_Float16*) alloc(SH * 2);
  _Float16*  vT16 = (_Float16*) alloc(SH * 2);
  _Float16*  a16  = (_Float16*) alloc(SH * 2);
  _Float16*  g16  = (_Float16*) alloc(SI * 2);
  _Float16*  wq16 = (_Float16*) alloc(NQ * 2);
  _Float16*  wk16 = (_Float16*) alloc(NQ * 2);
  _Float16*  wv16 = (_Float16*) alloc(NQ * 2);
  _Float16*  wo16 = (_Float16*) alloc(NQ * 2);
  _Float16*  wg16 = (_Float16*) alloc(NGU * 2);
  _Float16*  wu16 = (_Float16*) alloc(NGU * 2);
  _Float16*  wd16 = (_Float16*) alloc(NGU * 2);
  (void)off; (void)ws_size; (void)in_sizes; (void)n_in; (void)out_size;

  // residual stream starts as hidden_states
  (void)hipMemcpyAsync(x, hs, SH * 4, hipMemcpyDeviceToDevice, stream);

  // weight precasts (f32 -> f16), once per launch, deterministic
  cast_f16_kernel<<<4096, 256, 0, stream>>>(wq, wq16, NQ);
  cast_f16_kernel<<<4096, 256, 0, stream>>>(wk, wk16, NQ);
  cast_f16_kernel<<<4096, 256, 0, stream>>>(wv, wv16, NQ);
  cast_f16_kernel<<<4096, 256, 0, stream>>>(wo, wo16, NQ);
  cast_f16_kernel<<<4096, 256, 0, stream>>>(wg, wg16, NGU);
  cast_f16_kernel<<<4096, 256, 0, stream>>>(wu, wu16, NGU);
  cast_f16_kernel<<<4096, 256, 0, stream>>>(wd, wd16, NGU);

  dim3 gH(H_ / BN, S_ / BM);   // N=2048 gemms
  dim3 gI(I_ / BN, S_ / BM);   // N=5632 gemms

  for (int i = 0; i < L_; ++i) {
    const size_t oq = (size_t)i * H_ * H_;
    const size_t og = (size_t)i * I_ * H_;

    rmsnorm_kernel<<<S_, 256, 0, stream>>>(x, ln1 + i * H_, h16);

    gemm_f16_kernel<<<gH, 256, 0, stream>>>(h16, wq16 + oq, H_, q16, nullptr, nullptr, nullptr, 0);
    gemm_f16_kernel<<<gH, 256, 0, stream>>>(h16, wk16 + oq, H_, k16, nullptr, nullptr, nullptr, 0);
    gemm_f16_kernel<<<gH, 256, 0, stream>>>(h16, wv16 + oq, H_, vT16, nullptr, nullptr, nullptr, 4);

    rope_kernel<<<(S_ * NH_ * (D_ / 2)) / 256, 256, 0, stream>>>(q16, k16, cosT, sinT);

    flash_attn_kernel<<<(NH_ * (S_ / 16)) / 4, 128, 0, stream>>>(q16, k16, vT16, a16);

    gemm_f16_kernel<<<gH, 256, 0, stream>>>(a16, wo16 + oq, H_, nullptr, x, nullptr, x, 3);

    rmsnorm_kernel<<<S_, 256, 0, stream>>>(x, ln2 + i * H_, h16);

    gemm_f16_kernel<<<gI, 256, 0, stream>>>(h16, wg16 + og, H_, g16, nullptr, nullptr, nullptr, 1);
    gemm_f16_kernel<<<gI, 256, 0, stream>>>(h16, wu16 + og, H_, g16, nullptr, g16, nullptr, 2);
    gemm_f16_kernel<<<gH, 256, 0, stream>>>(g16, wd16 + og, I_, nullptr, x, nullptr, x, 3);
  }

  (void)hipMemcpyAsync(d_out, x, SH * 4, hipMemcpyDeviceToDevice, stream);
}
